// VisionSdpaAttention_48103633715762
// MI455X (gfx1250) — matrix operand order repaired and checked by reading, not compiled, not run on hardware
//
#include <hip/hip_runtime.h>

// ---------------------------------------------------------------------------
// GQA attention layer for MI455X (gfx1250, wave32, WMMA f32_16x16x32_f16)
//   q = x@Wq+bq ; k = x@Wk+bk ; v = x@Wv+bv ; rope(q,k) ; GQA flash attention ;
//   out = attn @ Wo
// Shapes: b=2, s=2048, MODEL_DIM=1024, KV_DIM=256, heads=16, kv_heads=4, d=64
//
// CDNA5-specific paths used:
//  - v_wmma_f32_16x16x32_f16 for all four matmul stages
//  - Tensor Data Mover (tensor_load_to_lds) double-buffered B-tile staging,
//    synchronized with s_wait_tensorcnt (shared weight tile loaded once per
//    block instead of once per wave)
//  - global_prefetch for the streamed K/V tiles in the attention kernel
// ---------------------------------------------------------------------------

#define SEQ      2048
#define MROWS    4096            // b * s
#define MD       1024            // model dim
#define KVD      256             // kv dim
#define NH       16
#define NKVH     4
#define HD       64

typedef __attribute__((ext_vector_type(8)))  _Float16 v8h;
typedef __attribute__((ext_vector_type(16))) _Float16 v16h;
typedef __attribute__((ext_vector_type(8)))  float    v8f;
typedef unsigned int u32x4 __attribute__((ext_vector_type(4)));
typedef int          i32x4 __attribute__((ext_vector_type(4)));
typedef int          i32x8 __attribute__((ext_vector_type(8)));

__device__ __forceinline__ v8f wmma_f16(v16h a, v16h b, v8f c) {
  return __builtin_amdgcn_wmma_f32_16x16x32_f16(
      /*neg_a=*/false, a, /*neg_b=*/false, b,
      /*c_mod=*/(short)0, c, /*reuse_a=*/false, /*reuse_b=*/false);
}

// A-frag: 16x32 tile from row-major data (ld in elements).
// lane holds row = lane%16; K = [(lane/16)*16, +16).
__device__ __forceinline__ v16h load_a_frag(const _Float16* __restrict__ base,
                                            int ld, int lane) {
  int row = lane & 15;
  int kb  = (lane >> 4) * 16;
  const _Float16* p = base + (size_t)row * ld + kb;
  v8h lo = *(const v8h*)p;
  v8h hi = *(const v8h*)(p + 8);
  v16h a;
#pragma unroll
  for (int i = 0; i < 8; ++i) { a[i] = lo[i]; a[i + 8] = hi[i]; }
  return a;
}

// B-frag: 32x16 (KxN) tile from a TRANSPOSED (NxK row-major) array in memory.
__device__ __forceinline__ v16h load_b_frag(const _Float16* __restrict__ bt,
                                            int ldk, int lane) {
  int n  = lane & 15;
  int kb = (lane >> 4) * 16;
  const _Float16* p = bt + (size_t)n * ldk + kb;
  v8h lo = *(const v8h*)p;
  v8h hi = *(const v8h*)(p + 8);
  v16h b;
#pragma unroll
  for (int i = 0; i < 8; ++i) { b[i] = lo[i]; b[i + 8] = hi[i]; }
  return b;
}

// B-frag from an LDS-staged tile (rows of 32 f16, unpadded, TDM row-major fill)
__device__ __forceinline__ v16h load_b_frag_lds(const _Float16* bsm, int lane) {
  int n  = lane & 15;
  int kb = (lane >> 4) * 16;
  const _Float16* p = bsm + n * 32 + kb;
  v8h lo = *(const v8h*)p;
  v8h hi = *(const v8h*)(p + 8);
  v16h b;
#pragma unroll
  for (int i = 0; i < 8; ++i) { b[i] = lo[i]; b[i + 8] = hi[i]; }
  return b;
}

// ---------------------------------------------------------------------------
// Tensor Data Mover: DMA a 64-row x 32-col f16 tile of a (nRows x 1024) f16
// row-major tensor into LDS (packed row-major, 4 KB). Issued once per wave;
// EXEC is ignored for TDM. Tracked by TENSORcnt.
// D# layout per CDNA5 ISA ch.8 (group0: 128b, group1: 256b).
// ---------------------------------------------------------------------------
__device__ __forceinline__ void tdm_load_tile(unsigned lds_addr,
                                              const _Float16* gptr,
                                              int nRows) {
  unsigned long long ga = (unsigned long long)(__UINTPTR_TYPE__)gptr;
  u32x4 g0;
  g0.x = 1u;                                          // count=1, user mode
  g0.y = lds_addr;                                    // lds_addr [63:32]
  g0.z = (unsigned)(ga & 0xFFFFFFFFu);                // global_addr[31:0]
  g0.w = (unsigned)((ga >> 32) & 0x01FFFFFFu)         // global_addr[56:32]
       | (2u << 30);                                  // type=2 ("image")
  i32x8 g1;
  g1[0] = 0x00010000;                 // workgroup_mask=0, data_size=1 (2B)
  g1[1] = (int)(MD << 16);            // tensor_dim0 = 1024 (bits 79:48 lo16)
  g1[2] = (int)((nRows & 0xFFFF) << 16); // tensor_dim0 hi=0, tensor_dim1 lo16
  g1[3] = (int)(32 << 16);            // tensor_dim1 hi=0, tile_dim0 = 32
  g1[4] = 64;                         // tile_dim1 = 64, tile_dim2 = 0
  g1[5] = MD;                         // tensor_dim0_stride = 1024 (lo32)
  g1[6] = 0;                          // stride hi, tensor_dim1_stride lo
  g1[7] = 0;
  i32x4 z4 = {0, 0, 0, 0};
#if __clang_major__ >= 23
  i32x8 z8 = {0, 0, 0, 0, 0, 0, 0, 0};
  __builtin_amdgcn_tensor_load_to_lds(g0, g1, z4, z4, z8, 0);
#else
  __builtin_amdgcn_tensor_load_to_lds(g0, g1, z4, z4, 0);
#endif
}

__device__ __forceinline__ unsigned lds_offset_of(const void* p) {
  // generic pointer to __shared__ -> low 32 bits are the LDS byte offset
  return (unsigned)(__UINTPTR_TYPE__)p;
}

// ---------------------------------------------------------------------------
// elementwise converts
// ---------------------------------------------------------------------------
__global__ void cvt_f32_to_f16_kernel(const float* __restrict__ src,
                                      _Float16* __restrict__ dst, int n) {
  int i = blockIdx.x * blockDim.x + threadIdx.x;
  if (i < n) dst[i] = (_Float16)src[i];
}

// src: K x N row-major f32 -> dst: N x K row-major f16 (coalesced writes)
__global__ void transpose_f16_kernel(const float* __restrict__ src,
                                     _Float16* __restrict__ dst, int K, int N) {
  int i = blockIdx.x * blockDim.x + threadIdx.x;
  if (i < N * K) {
    int n = i / K;
    int k = i - n * K;
    dst[i] = (_Float16)src[(size_t)k * N + n];
  }
}

// ---------------------------------------------------------------------------
// QKV projection GEMM, fused bias + RoPE + layout change.
// All 4 waves of a block share one 64-col weight slice -> B tile staged once
// per block via TDM into double-buffered LDS; WMMAs overlap the next DMA.
// mode 0: Q (rope, (b,h,s,d))  mode 1: K (rope, (b,kvh,s,d))
// mode 2: V (no rope, (b,kvh,d,s) transposed)
// ---------------------------------------------------------------------------
__global__ void qkv_gemm_kernel(const _Float16* __restrict__ Xh,
                                const _Float16* __restrict__ Wt,
                                const float* __restrict__ bias,
                                _Float16* __restrict__ Out,
                                int N, int nHeads, int mode) {
  const int K = MD;
  __shared__ __align__(16) _Float16 Bsm[2][64 * 32];  // 2 x 4 KB

  int lane = threadIdx.x & 31;
  int wv   = threadIdx.x >> 5;
  int m0   = blockIdx.x * 64 + wv * 16;  // 4 waves stacked in M
  int n0   = blockIdx.y * 64;            // one head's 64 columns

  if (wv == 0)
    tdm_load_tile(lds_offset_of(&Bsm[0][0]), Wt + (size_t)n0 * K, N);

  v8f acc[4] = {};
  int buf = 0;
  for (int k0 = 0; k0 < K; k0 += 32) {
    if (wv == 0) {
      if (k0 + 32 < K) {
        tdm_load_tile(lds_offset_of(&Bsm[buf ^ 1][0]),
                      Wt + (size_t)n0 * K + (k0 + 32), N);
        __builtin_amdgcn_s_wait_tensorcnt(1);  // in-order: current tile done
      } else {
        __builtin_amdgcn_s_wait_tensorcnt(0);
      }
    }
    __syncthreads();

    v16h a = load_a_frag(Xh + (size_t)m0 * K + k0, K, lane);
#pragma unroll
    for (int t = 0; t < 4; ++t) {
      v16h b = load_b_frag_lds(&Bsm[buf][t * 16 * 32], lane);
      acc[t] = wmma_f16(a, b, acc[t]);
    }

    __syncthreads();  // everyone done with Bsm[buf] before it is re-filled
    buf ^= 1;
  }

  int ln = lane & 15, hf = lane >> 4;
#pragma unroll
  for (int t = 0; t < 4; ++t) {
    float bv = bias[n0 + t * 16 + ln];
#pragma unroll
    for (int r = 0; r < 8; ++r) acc[t][r] += bv;
  }

  int b_   = m0 >> 11;                  // batch
  int s0   = (m0 & (SEQ - 1)) + hf * 8; // sequence position base (+r)
  int head = n0 >> 6;

  if (mode != 2) {
    // fused RoPE: per-lane, pairs (t, t+2) == dims (d, d+32), half = 32
#pragma unroll
    for (int t = 0; t < 2; ++t) {
      int   d   = t * 16 + ln;
      float inv = __powf(10000.0f, -(float)d * (1.0f / 32.0f));
#pragma unroll
      for (int r = 0; r < 8; ++r) {
        float ang = (float)(s0 + r) * inv;
        float c = __cosf(ang), sn = __sinf(ang);
        float x1 = acc[t][r], x2 = acc[t + 2][r];
        acc[t][r]     = x1 * c - x2 * sn;
        acc[t + 2][r] = x2 * c + x1 * sn;
      }
    }
#pragma unroll
    for (int t = 0; t < 4; ++t) {
      int d = t * 16 + ln;
      size_t base = ((size_t)(b_ * nHeads + head) * SEQ) * HD + d;
#pragma unroll
      for (int r = 0; r < 8; ++r)
        Out[base + (size_t)(s0 + r) * HD] = (_Float16)acc[t][r];
    }
  } else {
    // V: store transposed (b, head, d, s) -> 8 consecutive s per lane
#pragma unroll
    for (int t = 0; t < 4; ++t) {
      int d = t * 16 + ln;
      v8h pk;
#pragma unroll
      for (int r = 0; r < 8; ++r) pk[r] = (_Float16)acc[t][r];
      *(v8h*)(Out + ((size_t)(b_ * nHeads + head) * HD + d) * SEQ + s0) = pk;
    }
  }
}

// ---------------------------------------------------------------------------
// Flash attention: 1 wave per 16-query tile, online softmax over 2048 keys in
// chunks of 32. Row reductions via __shfl_xor across the 16-lane half-waves.
// P staged through LDS (C-layout -> A-layout). V pre-transposed (d, s).
// ---------------------------------------------------------------------------
__global__ void attn_kernel(const _Float16* __restrict__ Qh,  // (b,h,s,d)
                            const _Float16* __restrict__ Kh,  // (b,kvh,s,d)
                            const _Float16* __restrict__ Vt,  // (b,kvh,d,s)
                            _Float16* __restrict__ AO) {      // (b,s,h*64+d)
  __shared__ __align__(16) _Float16 plds[4][16][40];  // 16x32 P tile, padded

  int lane = threadIdx.x & 31;
  int wv   = threadIdx.x >> 5;
  int bh   = blockIdx.x;          // b*16 + h
  int b_   = bh >> 4, h = bh & 15;
  int kvh  = h >> 2;              // GQA ratio 4
  int q0   = (blockIdx.y * 4 + wv) * 16;

  const _Float16* Qb = Qh + ((size_t)bh * SEQ + q0) * HD;
  const _Float16* Kb = Kh + ((size_t)(b_ * NKVH + kvh) * SEQ) * HD;
  const _Float16* Vb = Vt + ((size_t)(b_ * NKVH + kvh) * HD) * SEQ;

  v16h qa0 = load_a_frag(Qb, HD, lane);       // dims  0..31
  v16h qa1 = load_a_frag(Qb + 32, HD, lane);  // dims 32..63

  v8f o[4] = {};
  float m[8], l[8];
#pragma unroll
  for (int r = 0; r < 8; ++r) { m[r] = -1e30f; l[r] = 0.0f; }

  int ln = lane & 15, hf = lane >> 4;
  const float scale = 0.125f;  // 1/sqrt(64)

  for (int kv = 0; kv < SEQ; kv += 32) {
    // prefetch the next KV chunk while this one is consumed
    if (kv + 32 < SEQ) {
      __builtin_prefetch(Kb + (size_t)(kv + 32) * HD, 0, 1);
      __builtin_prefetch(Vb + (size_t)(kv + 32), 0, 1);
    }

    // S = Q @ K^T for 32 keys -> two 16x16 f32 tiles
    v8f s0 = {}, s1 = {};
    {
      v16h kb;
      kb = load_b_frag(Kb + (size_t)kv * HD, HD, lane);       // keys kv..+15, d 0..31
      s0 = wmma_f16(qa0, kb, s0);
      kb = load_b_frag(Kb + (size_t)kv * HD + 32, HD, lane);  // d 32..63
      s0 = wmma_f16(qa1, kb, s0);
      kb = load_b_frag(Kb + (size_t)(kv + 16) * HD, HD, lane);
      s1 = wmma_f16(qa0, kb, s1);
      kb = load_b_frag(Kb + (size_t)(kv + 16) * HD + 32, HD, lane);
      s1 = wmma_f16(qa1, kb, s1);
    }

    float al[8];
#pragma unroll
    for (int r = 0; r < 8; ++r) {
      float v0 = s0[r] * scale, v1 = s1[r] * scale;
      float mx = fmaxf(v0, v1);
      mx = fmaxf(mx, __shfl_xor(mx, 1));
      mx = fmaxf(mx, __shfl_xor(mx, 2));
      mx = fmaxf(mx, __shfl_xor(mx, 4));
      mx = fmaxf(mx, __shfl_xor(mx, 8));
      float mn = fmaxf(m[r], mx);
      al[r] = __expf(m[r] - mn);
      m[r] = mn;
      float p0 = __expf(v0 - mn);
      float p1 = __expf(v1 - mn);
      s0[r] = p0; s1[r] = p1;
      float rs = p0 + p1;
      rs += __shfl_xor(rs, 1);
      rs += __shfl_xor(rs, 2);
      rs += __shfl_xor(rs, 4);
      rs += __shfl_xor(rs, 8);
      l[r] = l[r] * al[r] + rs;
    }

    // rescale running O
#pragma unroll
    for (int t = 0; t < 4; ++t)
#pragma unroll
      for (int r = 0; r < 8; ++r) o[t][r] *= al[r];

    // C-layout -> LDS (row-major 16x32, stride 40) -> A-layout
#pragma unroll
    for (int r = 0; r < 8; ++r) {
      int row = r + 8 * hf;
      plds[wv][row][ln]      = (_Float16)s0[r];
      plds[wv][row][16 + ln] = (_Float16)s1[r];
    }
    v16h pa = load_a_frag(&plds[wv][0][0], 40, lane);

    // O += P @ V
#pragma unroll
    for (int t = 0; t < 4; ++t) {
      v16h vb = load_b_frag(Vb + (size_t)(t * 16) * SEQ + kv, SEQ, lane);
      o[t] = wmma_f16(pa, vb, o[t]);
    }
  }

  // normalize + store to (b, s, h*64+d) f16 for the output projection
#pragma unroll
  for (int t = 0; t < 4; ++t) {
    int d = t * 16 + ln;
#pragma unroll
    for (int r = 0; r < 8; ++r) {
      int srow = q0 + r + 8 * hf;
      AO[((size_t)b_ * SEQ + srow) * MD + h * HD + d] = (_Float16)(o[t][r] / l[r]);
    }
  }
}

// ---------------------------------------------------------------------------
// Output projection: out = AO @ Wo (f32), TDM-staged B tiles like qkv_gemm.
// ---------------------------------------------------------------------------
__global__ void out_gemm_kernel(const _Float16* __restrict__ Ah,   // 4096x1024
                                const _Float16* __restrict__ Wot,  // 1024x1024 (NxK)
                                float* __restrict__ Out) {
  const int K = MD, N = MD;
  __shared__ __align__(16) _Float16 Bsm[2][64 * 32];

  int lane = threadIdx.x & 31;
  int wv   = threadIdx.x >> 5;
  int m0   = blockIdx.x * 64 + wv * 16;
  int n0   = blockIdx.y * 64;

  if (wv == 0)
    tdm_load_tile(lds_offset_of(&Bsm[0][0]), Wot + (size_t)n0 * K, N);

  v8f acc[4] = {};
  int buf = 0;
  for (int k0 = 0; k0 < K; k0 += 32) {
    if (wv == 0) {
      if (k0 + 32 < K) {
        tdm_load_tile(lds_offset_of(&Bsm[buf ^ 1][0]),
                      Wot + (size_t)n0 * K + (k0 + 32), N);
        __builtin_amdgcn_s_wait_tensorcnt(1);
      } else {
        __builtin_amdgcn_s_wait_tensorcnt(0);
      }
    }
    __syncthreads();

    v16h a = load_a_frag(Ah + (size_t)m0 * K + k0, K, lane);
#pragma unroll
    for (int t = 0; t < 4; ++t) {
      v16h b = load_b_frag_lds(&Bsm[buf][t * 16 * 32], lane);
      acc[t] = wmma_f16(a, b, acc[t]);
    }

    __syncthreads();
    buf ^= 1;
  }

  int ln = lane & 15, hf = lane >> 4;
#pragma unroll
  for (int t = 0; t < 4; ++t)
#pragma unroll
    for (int r = 0; r < 8; ++r)
      Out[(size_t)(m0 + r + 8 * hf) * N + n0 + t * 16 + ln] = acc[t][r];
}

// ---------------------------------------------------------------------------
extern "C" void kernel_launch(void* const* d_in, const int* in_sizes, int n_in,
                              void* d_out, int out_size, void* d_ws, size_t ws_size,
                              hipStream_t stream) {
  (void)in_sizes; (void)n_in; (void)out_size; (void)ws_size;
  const float* x  = (const float*)d_in[0];
  const float* Wq = (const float*)d_in[1];
  const float* bq = (const float*)d_in[2];
  const float* Wk = (const float*)d_in[3];
  const float* bk = (const float*)d_in[4];
  const float* Wv = (const float*)d_in[5];
  const float* bv = (const float*)d_in[6];
  const float* Wo = (const float*)d_in[7];
  float* out = (float*)d_out;

  char* ws = (char*)d_ws;
  size_t off = 0;
  auto alloc = [&](size_t bytes) -> void* {
    void* p = ws + off;
    off += (bytes + 255) & ~(size_t)255;
    return p;
  };
  _Float16* Xh  = (_Float16*)alloc((size_t)MROWS * MD * 2);
  _Float16* Wqt = (_Float16*)alloc((size_t)MD * MD * 2);
  _Float16* Wkt = (_Float16*)alloc((size_t)KVD * MD * 2);
  _Float16* Wvt = (_Float16*)alloc((size_t)KVD * MD * 2);
  _Float16* Wot = (_Float16*)alloc((size_t)MD * MD * 2);
  _Float16* Qh  = (_Float16*)alloc((size_t)2 * NH * SEQ * HD * 2);
  _Float16* Kh  = (_Float16*)alloc((size_t)2 * NKVH * SEQ * HD * 2);
  _Float16* Vtb = (_Float16*)alloc((size_t)2 * NKVH * HD * SEQ * 2);
  _Float16* AO  = (_Float16*)alloc((size_t)MROWS * MD * 2);

  // 1) converts / transposes
  {
    int n = MROWS * MD;
    cvt_f32_to_f16_kernel<<<(n + 255) / 256, 256, 0, stream>>>(x, Xh, n);
    int nq = MD * MD;
    transpose_f16_kernel<<<(nq + 255) / 256, 256, 0, stream>>>(Wq, Wqt, MD, MD);
    int nk = MD * KVD;
    transpose_f16_kernel<<<(nk + 255) / 256, 256, 0, stream>>>(Wk, Wkt, MD, KVD);
    transpose_f16_kernel<<<(nk + 255) / 256, 256, 0, stream>>>(Wv, Wvt, MD, KVD);
    transpose_f16_kernel<<<(nq + 255) / 256, 256, 0, stream>>>(Wo, Wot, MD, MD);
  }

  // 2) QKV projections with fused bias + RoPE + layout, TDM-staged weights
  qkv_gemm_kernel<<<dim3(MROWS / 64, MD / 64),  128, 0, stream>>>(Xh, Wqt, bq, Qh,  MD,  NH,   0);
  qkv_gemm_kernel<<<dim3(MROWS / 64, KVD / 64), 128, 0, stream>>>(Xh, Wkt, bk, Kh,  KVD, NKVH, 1);
  qkv_gemm_kernel<<<dim3(MROWS / 64, KVD / 64), 128, 0, stream>>>(Xh, Wvt, bv, Vtb, KVD, NKVH, 2);

  // 3) flash attention: 32 (b,h) x 32 query-tile-groups, 4 waves per block
  attn_kernel<<<dim3(2 * NH, SEQ / 64), 128, 0, stream>>>(Qh, Kh, Vtb, AO);

  // 4) output projection
  out_gemm_kernel<<<dim3(MROWS / 64, MD / 64), 128, 0, stream>>>(AO, Wot, out);
}
